// HadamardMultiplier_62740882260479
// MI455X (gfx1250) — matrix operand order repaired and checked by reading, not compiled, not run on hardware
//
#include <hip/hip_runtime.h>
#include <hip/hip_bf16.h>

// ---------------------------------------------------------------------------
// Problem constants (from the reference)
// ---------------------------------------------------------------------------
static constexpr int  KDIM  = 172;            // number of 64-blocks
static constexpr int  SUBL  = 64;             // FHT length
static constexpr int  NDIM  = KDIM * SUBL;    // 11008
static constexpr int  TDIM  = 8192;
static constexpr int  DOUT  = 4096;
static constexpr int  KP    = 192;            // KDIM padded to 6*32 for WMMA K-loop
static constexpr long DMCOL = (long)DOUT * SUBL; // 262144 columns of the mix GEMM
static constexpr float INV_SQRT_N = 0.009531165f; // 1/sqrt(11008)

typedef __attribute__((ext_vector_type(16))) __bf16 v16bf;
typedef __attribute__((ext_vector_type(8)))  __bf16 v8bf;
typedef __attribute__((ext_vector_type(8)))  float  v8f;

__device__ __forceinline__ unsigned short f2bf(float f) {
  // round-to-nearest-even fp32 -> bf16
  unsigned u = __float_as_uint(f);
  u += 0x7FFFu + ((u >> 16) & 1u);
  return (unsigned short)(u >> 16);
}

__device__ __forceinline__ unsigned pk2(float a, float b) {
  return (unsigned)f2bf(a) | ((unsigned)f2bf(b) << 16);
}

__device__ __forceinline__ v16bf mk16(v8bf lo, v8bf hi) {
  v16bf r;
#pragma unroll
  for (int i = 0; i < 8; ++i) { r[i] = lo[i]; r[i + 8] = hi[i]; }
  return r;
}

// 32-byte per-lane async copy global -> LDS (ASYNCcnt path, CDNA5).
// INST_OFFSET is added to BOTH the LDS dest and the global address (ISA
// 08_async_tensor 4.4), so offset:16 covers the second b128 of the chunk.
__device__ __forceinline__ void async_cp32(unsigned lds, const void* g) {
  asm volatile(
      "global_load_async_to_lds_b128 %0, %1, off\n\t"
      "global_load_async_to_lds_b128 %0, %1, off offset:16"
      :: "v"(lds), "v"(g) : "memory");
}

__device__ __forceinline__ void wait_async0() {
#if __has_builtin(__builtin_amdgcn_s_wait_asynccnt)
  __builtin_amdgcn_s_wait_asynccnt(0);
#else
  asm volatile("s_wait_asynccnt 0x0" ::: "memory");
#endif
}

// ---------------------------------------------------------------------------
// K0: had_K (172x172 f32, +-1) -> hadb (192x192 bf16), TRANSPOSED + zero pad.
// hadb[j][k] = had[k][j]; bf16 is exact for +-1.
// ---------------------------------------------------------------------------
__global__ void k_prep_had(const float* __restrict__ had,
                           unsigned short* __restrict__ hadb) {
  int idx = blockIdx.x * 256 + threadIdx.x;
  if (idx >= KP * KP) return;
  int j = idx / KP, k = idx % KP;
  unsigned short v = 0;
  if (j < KDIM && k < KDIM) v = f2bf(had[k * KDIM + j]);
  hadb[idx] = v;
}

// ---------------------------------------------------------------------------
// K1: x f32 -> bf16 (A matrix of the main GEMM), 8 elements / thread.
// ---------------------------------------------------------------------------
__global__ void k_cvt_x(const float* __restrict__ x,
                        unsigned short* __restrict__ xb, long n) {
  long i = ((long)blockIdx.x * 256 + threadIdx.x) * 8;
  if (i >= n) return;
  float4 a = *(const float4*)(x + i);
  float4 b = *(const float4*)(x + i + 4);
  uint4 p;
  p.x = pk2(a.x, a.y); p.y = pk2(a.z, a.w);
  p.z = pk2(b.x, b.y); p.w = pk2(b.z, b.w);
  *(uint4*)(xb + i) = p;
}

// ---------------------------------------------------------------------------
// K2: per-row, per-64-block FHT of W (in registers) * 1/sqrt(N), write bf16
// in layout wf[k][d*64+m]  (k = block index = reduction index of mix GEMM).
// One thread owns one (d,k) 64-block.
// ---------------------------------------------------------------------------
__global__ void k_fht_w(const float* __restrict__ W,
                        unsigned short* __restrict__ wf) {
  int idx = blockIdx.x * 256 + threadIdx.x;
  if (idx >= DOUT * KDIM) return;
  int d = idx / KDIM, k = idx % KDIM;
  const float* src = W + (long)d * NDIM + k * SUBL;
  float v[64];
#pragma unroll
  for (int i = 0; i < 16; ++i) {
    float4 t = *(const float4*)(src + i * 4);
    v[4 * i + 0] = t.x; v[4 * i + 1] = t.y;
    v[4 * i + 2] = t.z; v[4 * i + 3] = t.w;
  }
#pragma unroll
  for (int h = 1; h < 64; h <<= 1) {
#pragma unroll
    for (int i = 0; i < 64; ++i) {
      if ((i & h) == 0) {
        float a = v[i], b = v[i + h];
        v[i] = a + b; v[i + h] = a - b;
      }
    }
  }
  unsigned short* dst = wf + (long)k * DMCOL + (long)d * SUBL;
#pragma unroll
  for (int i = 0; i < 8; ++i) {
    uint4 p;
    p.x = pk2(v[8 * i + 0] * INV_SQRT_N, v[8 * i + 1] * INV_SQRT_N);
    p.y = pk2(v[8 * i + 2] * INV_SQRT_N, v[8 * i + 3] * INV_SQRT_N);
    p.z = pk2(v[8 * i + 4] * INV_SQRT_N, v[8 * i + 5] * INV_SQRT_N);
    p.w = pk2(v[8 * i + 6] * INV_SQRT_N, v[8 * i + 7] * INV_SQRT_N);
    *(uint4*)(dst + 8 * i) = p;
  }
}

// ---------------------------------------------------------------------------
// K3: mix GEMM via WMMA:  wt[j][dm] = sum_k hadb[j][k] * wf[k][dm]
//   M = 192 (rows j, padded; only j<172 stored), N-tile = 128 cols (dm),
//   K-loop = 6 steps of 32. A-frags straight from global (hadb is 72KB,
//   cache resident). B staged transposed in LDS -> B-frags are 2x b128.
// Output scattered into wt[d][j*64+m] (bf16 B matrix of the main GEMM).
// ---------------------------------------------------------------------------
__global__ __launch_bounds__(256) void k_mix_w(
    const unsigned short* __restrict__ hadb_,
    const unsigned short* __restrict__ wf_,
    unsigned short* __restrict__ wt) {
  __shared__ __align__(16) __bf16 Bs[128 * 32];   // [col][j']  8 KB
  const __bf16* hadb = (const __bf16*)hadb_;
  const __bf16* wf   = (const __bf16*)wf_;

  const int tid  = threadIdx.x;
  const int lane = tid & 31;
  const int w    = tid >> 5;
  const long c0  = (long)blockIdx.x * 128;
  const int Mbase = (w & 3) * 48;   // 4 wave rows * 48 = 192
  const int Nbase = (w >> 2) * 64;  // 2 wave cols * 64 = 128

  v8f zero = {};
  v8f acc[3][4];
#pragma unroll
  for (int mt = 0; mt < 3; ++mt)
#pragma unroll
    for (int nt = 0; nt < 4; ++nt) acc[mt][nt] = zero;

  const int jrow = tid >> 3;        // 0..31 : j' within k-chunk
  const int colb = (tid & 7) * 16;  // 0..112

  for (int kc = 0; kc < 6; ++kc) {
    // ---- stage B chunk (32 x 128), transposed into LDS, zero pad j>=172
    int jg = kc * 32 + jrow;
    v8bf lo = {}, hi = {};
    if (jg < KDIM) {
      const __bf16* p = wf + (long)jg * DMCOL + c0 + colb;
      lo = *(const v8bf*)p;
      hi = *(const v8bf*)(p + 8);
    }
#pragma unroll
    for (int i = 0; i < 8; ++i) {
      Bs[(colb + i)     * 32 + jrow] = lo[i];
      Bs[(colb + 8 + i) * 32 + jrow] = hi[i];
    }
    __syncthreads();

    // ---- A-frags from global
    v16bf afr[3];
#pragma unroll
    for (int mt = 0; mt < 3; ++mt) {
      int row = Mbase + mt * 16 + (lane & 15);
      const __bf16* p = hadb + row * KP + kc * 32 + (lane >> 4) * 16;
      afr[mt] = mk16(*(const v8bf*)p, *(const v8bf*)(p + 8));
    }
    // ---- B-frags from LDS + WMMA
#pragma unroll
    for (int nt = 0; nt < 4; ++nt) {
      int col = Nbase + nt * 16 + (lane & 15);
      const __bf16* q = &Bs[col * 32 + (lane >> 4) * 16];
      v16bf bfr = mk16(*(const v8bf*)q, *(const v8bf*)(q + 8));
#pragma unroll
      for (int mt = 0; mt < 3; ++mt)
        acc[mt][nt] = __builtin_amdgcn_wmma_f32_16x16x32_bf16(
            false, afr[mt], false, bfr, (short)0, acc[mt][nt], false, false);
    }
    __syncthreads();
  }

  // ---- epilogue: scatter bf16 into wt[d*NDIM + j*64 + m]
#pragma unroll
  for (int mt = 0; mt < 3; ++mt) {
    int jb = Mbase + mt * 16 + 8 * (lane >> 4);
#pragma unroll
    for (int nt = 0; nt < 4; ++nt) {
      long dm = c0 + Nbase + nt * 16 + (lane & 15);
      unsigned short* dst = wt + (dm >> 6) * NDIM + (dm & 63);
#pragma unroll
      for (int r = 0; r < 8; ++r) {
        int j = jb + r;
        if (j < KDIM) dst[(long)j * 64] = f2bf(acc[mt][nt][r]);
      }
    }
  }
}

// ---------------------------------------------------------------------------
// K4: main GEMM  out[t][d] = sum_n xb[t][n]*wt[d][n] + b[d]
//   128x128 tile / workgroup, 8 waves (each 64x32 -> 8 accum tiles),
//   k-step 32, double-buffered LDS (2 x 16 KB), GLOBAL_LOAD_ASYNC_TO_LDS_B128
//   staging tracked with ASYNCcnt, one s_wait_asynccnt + barrier per k-step.
// ---------------------------------------------------------------------------
__global__ __launch_bounds__(256) void k_gemm(
    const unsigned short* __restrict__ xb_,
    const unsigned short* __restrict__ wt_,
    const float* __restrict__ bias,
    float* __restrict__ out) {
  __shared__ __align__(16) __bf16 As[2][128 * 32];  // [t'][k]
  __shared__ __align__(16) __bf16 Bs[2][128 * 32];  // [d'][k]

  const int tid  = threadIdx.x;
  const int lane = tid & 31;
  const int w    = tid >> 5;
  const int t0 = (blockIdx.x >> 5) * 128;  // 64 t-tiles
  const int d0 = (blockIdx.x & 31) * 128;  // 32 d-tiles
  const int Mbase = (w >> 2) * 64;         // 2 wave rows
  const int Nbase = (w & 3) * 32;          // 4 wave cols

  // staging assignment: 256 threads x 32B cover one 128x32 bf16 tile
  const int arow = tid >> 1;
  const int acol = (tid & 1) * 16;
  const char* agp = (const char*)(xb_ + (long)(t0 + arow) * NDIM + acol);
  const char* bgp = (const char*)(wt_ + (long)(d0 + arow) * NDIM + acol);
  const int soff  = arow * 32 + acol;
  // LDS byte addresses (generic-pointer low 32 bits == LDS offset, ISA 10.2)
  unsigned lA[2] = { (unsigned)(unsigned long long)&As[0][soff],
                     (unsigned)(unsigned long long)&As[1][soff] };
  unsigned lB[2] = { (unsigned)(unsigned long long)&Bs[0][soff],
                     (unsigned)(unsigned long long)&Bs[1][soff] };

  v8f zero = {};
  v8f acc[4][2];
#pragma unroll
  for (int mt = 0; mt < 4; ++mt)
#pragma unroll
    for (int nt = 0; nt < 2; ++nt) acc[mt][nt] = zero;

  // prologue: chunk 0 -> LDS buffer 0 via async copy engine
  async_cp32(lA[0], agp);
  async_cp32(lB[0], bgp);
  wait_async0();
  __syncthreads();

  const int NK = NDIM / 32;  // 344
  for (int kc = 0; kc < NK; ++kc) {
    const int cur = kc & 1;

    // kick off next chunk into the other buffer (overlaps with WMMAs below);
    // safe: that buffer's readers finished before the barrier we just passed.
    if (kc + 1 < NK) {
      long ko = (long)(kc + 1) * 64;  // 32 bf16 = 64 bytes per k-step
      async_cp32(lA[cur ^ 1], agp + ko);
      async_cp32(lB[cur ^ 1], bgp + ko);
    }

    // A-frags (16x32 bf16): lanes 0-15 hold K{0..15}, lanes 16-31 K{16..31}
    v16bf afr[4];
#pragma unroll
    for (int mt = 0; mt < 4; ++mt) {
      const __bf16* p = &As[cur][(Mbase + mt * 16 + (lane & 15)) * 32 + (lane >> 4) * 16];
      afr[mt] = mk16(*(const v8bf*)p, *(const v8bf*)(p + 8));
    }
    // B-frags (32x16): lane holds 16 K-contiguous elems of one column (= wt row)
#pragma unroll
    for (int nt = 0; nt < 2; ++nt) {
      const __bf16* q = &Bs[cur][(Nbase + nt * 16 + (lane & 15)) * 32 + (lane >> 4) * 16];
      v16bf bfr = mk16(*(const v8bf*)q, *(const v8bf*)(q + 8));
#pragma unroll
      for (int mt = 0; mt < 4; ++mt)
        acc[mt][nt] = __builtin_amdgcn_wmma_f32_16x16x32_bf16(
            false, afr[mt], false, bfr, (short)0, acc[mt][nt], false, false);
    }

    // drain this wave's async copies, then make them visible to all waves
    if (kc + 1 < NK) {
      wait_async0();
      __syncthreads();
    }
  }

  // epilogue: f32 store + bias (each lane's tile column d is fixed)
#pragma unroll
  for (int nt = 0; nt < 2; ++nt) {
    int d = d0 + Nbase + nt * 16 + (lane & 15);
    float bv = bias[d];
#pragma unroll
    for (int mt = 0; mt < 4; ++mt) {
      int trow = t0 + Mbase + mt * 16 + 8 * (lane >> 4);
      float* o = out + (long)trow * DOUT + d;
#pragma unroll
      for (int r = 0; r < 8; ++r)
        o[(long)r * DOUT] = acc[mt][nt][r] + bv;
    }
  }
}

// ---------------------------------------------------------------------------
// Launch: inputs {x, had_K, W, b}. Workspace layout (bytes):
//   xb   : T*N*2      = 180,355,072
//   wt   : DOUT*N*2   =  90,177,536
//   wf   : K*DMCOL*2  =  90,177,536
//   hadb : 192*192*2  =      73,728    (total ~361 MB)
// ---------------------------------------------------------------------------
extern "C" void kernel_launch(void* const* d_in, const int* in_sizes, int n_in,
                              void* d_out, int out_size, void* d_ws, size_t ws_size,
                              hipStream_t stream) {
  const float* x   = (const float*)d_in[0];
  const float* had = (const float*)d_in[1];
  const float* W   = (const float*)d_in[2];
  const float* b   = (const float*)d_in[3];
  float* out = (float*)d_out;

  char* ws = (char*)d_ws;
  unsigned short* xb   = (unsigned short*)(ws);
  unsigned short* wt   = (unsigned short*)(ws + (size_t)TDIM * NDIM * 2);
  unsigned short* wf   = (unsigned short*)(ws + (size_t)TDIM * NDIM * 2
                                              + (size_t)DOUT * NDIM * 2);
  unsigned short* hadb = (unsigned short*)(ws + (size_t)TDIM * NDIM * 2
                                              + (size_t)DOUT * NDIM * 2
                                              + (size_t)KDIM * DMCOL * 2);

  k_prep_had<<<(KP * KP + 255) / 256, 256, 0, stream>>>(had, hadb);

  long nx = (long)TDIM * NDIM;
  k_cvt_x<<<(unsigned)((nx / 8 + 255) / 256), 256, 0, stream>>>(x, xb, nx);

  k_fht_w<<<(DOUT * KDIM + 255) / 256, 256, 0, stream>>>(W, wf);

  k_mix_w<<<(unsigned)(DMCOL / 128), 256, 0, stream>>>(hadb, wf, wt);

  k_gemm<<<(TDIM / 128) * (DOUT / 128), 256, 0, stream>>>(xb, wt, b, out);
}
